// HNestedUNet_44976897523755
// MI455X (gfx1250) — matrix operand order repaired and checked by reading, not compiled, not run on hardware
//
#include <hip/hip_runtime.h>
#include <cmath>

// ---------------------------------------------------------------------------
// CDNA5 (gfx1250, wave32) hyperbolic UNet++.
// 3x3 convs: implicit GEMM on V_WMMA_F32_16X16X32_BF16.
//  - Weights prepacked to the wave32 A-fragment layout, then streamed into
//    double-buffered LDS with GLOBAL_LOAD_ASYNC_TO_LDS_B128 (ASYNCcnt).
//  - K ordered (ky,kx,cin) with cin padded to 16 -> branchless clamped gather
//    of the B tile into double-buffered LDS shared by WM waves.
// ---------------------------------------------------------------------------

#define HSQC 0.31622776601683794f   // sqrt(0.1)

typedef __attribute__((ext_vector_type(16))) __bf16 v16bf;
typedef __attribute__((ext_vector_type(8)))  float  v8f;

__device__ __forceinline__ unsigned short f2bfu(float f) {
  union { float f; unsigned u; } c; c.f = f;
  unsigned u = c.u;
  u += 0x7fffu + ((u >> 16) & 1u);        // round-to-nearest-even
  return (unsigned short)(u >> 16);
}

// ---- logmap0: per-pixel channel-norm, artanh scaling (NCHW, N=2) ----------
__global__ void k_logmap(const float* __restrict__ in, float* __restrict__ out,
                         int C, int HW) {
  int p = blockIdx.x * blockDim.x + threadIdx.x;
  int NHW = 2 * HW;
  if (p >= NHW) return;
  int n = p / HW, rem = p - n * HW;
  const float* ip = in + (size_t)n * C * HW + rem;
  float ss = 0.f;
  for (int c = 0; c < C; ++c) { float v = ip[(size_t)c * HW]; ss += v * v; }
  float nn = fmaxf(sqrtf(ss), 1e-5f);
  float a  = fminf(fmaxf(HSQC * nn, 1e-5f), 1.f - 1e-5f);
  float s  = atanhf(a) / a;
  float* op = out + (size_t)n * C * HW + rem;
  for (int c = 0; c < C; ++c) op[(size_t)c * HW] = s * ip[(size_t)c * HW];
}

// ---- expmap0 (optionally after ReLU in tangent space) ---------------------
__global__ void k_manifold(const float* __restrict__ in, float* __restrict__ out,
                           int C, int HW, int relu) {
  int p = blockIdx.x * blockDim.x + threadIdx.x;
  int NHW = 2 * HW;
  if (p >= NHW) return;
  int n = p / HW, rem = p - n * HW;
  const float* ip = in + (size_t)n * C * HW + rem;
  float ss = 0.f;
  for (int c = 0; c < C; ++c) {
    float v = ip[(size_t)c * HW];
    if (relu) v = fmaxf(v, 0.f);
    ss += v * v;
  }
  float nn = fmaxf(sqrtf(ss), 1e-5f);
  float s  = tanhf(HSQC * nn) / (HSQC * nn);
  float* op = out + (size_t)n * C * HW + rem;
  for (int c = 0; c < C; ++c) {
    float v = ip[(size_t)c * HW];
    if (relu) v = fmaxf(v, 0.f);
    op[(size_t)c * HW] = s * v;
  }
}

// ---- weight prepack: OIHW f32 -> bf16 wave32 A-fragment layout ------------
// K ordering: k = r*Cp + cin  (r = ky*3+kx, Cp = Cin padded to 16).
// element i: e=i&15, lane=(i>>4)&31, chunk=(i>>9)%nkc, mtile=(i>>9)/nkc
// A 16x32 layout: lane L holds row M=L%16; K = chunk*32+(L/16)*16+e
__global__ void k_pack_w(const float* __restrict__ Wsrc,
                         unsigned short* __restrict__ dst,
                         int Cout, int Cin) {
  int Cp  = (Cin + 15) & ~15;
  int nkc = (9 * Cp + 31) / 32;
  size_t total = (size_t)(Cout >> 4) * nkc * 512;
  for (size_t i = (size_t)blockIdx.x * blockDim.x + threadIdx.x; i < total;
       i += (size_t)gridDim.x * blockDim.x) {
    int e    = (int)(i & 15);
    int lane = (int)((i >> 4) & 31);
    size_t rest = i >> 9;
    int c  = (int)(rest % nkc);
    int mt = (int)(rest / nkc);
    int M = mt * 16 + (lane & 15);
    int k = c * 32 + (lane >> 4) * 16 + e;
    int r = k / Cp, cin = k - r * Cp;
    float v = (r < 9 && cin < Cin)
                  ? Wsrc[(size_t)M * Cin * 9 + (size_t)cin * 9 + r] : 0.f;
    dst[i] = f2bfu(v);
  }
}

// ---- 3x3 conv (pad 1) as implicit GEMM on WMMA ----------------------------
// WM waves share one 16-pixel B tile (WM = min(Cout/16, 8)); WN = 8/WM tiles
// per block. A tiles stream global->LDS via async-DMA; B tiles are gathered
// cooperatively into LDS. Both double-buffered.
template <int WM>
__global__ void __launch_bounds__(256)
k_conv3x3_wmma(const float* __restrict__ vin,
               const unsigned short* __restrict__ wpack,
               const float* __restrict__ bias,
               float* __restrict__ out,
               int Cin, int Cout, int H, int W) {
  constexpr int WN  = 8 / WM;
  constexpr int EPW = 16 / WM;               // staged B elements per lane
  const int HW  = H * W;
  const int NHW = 2 * HW;
  const int Cp  = (Cin + 15) & ~15;
  const int nkc = (9 * Cp + 31) / 32;
  const int mtiles = Cout >> 4;
  const int ntiles = (NHW + 15) >> 4;
  const int ngroups = (ntiles + WN - 1) / WN;

  __shared__ __align__(32) unsigned short sA[2][WM][512];
  __shared__ __align__(32) unsigned short sB[2][WN][512];

  int w    = threadIdx.x >> 5;
  int lane = threadIdx.x & 31;
  int wm = w % WM, wn = w / WM;
  int mg = blockIdx.x / ngroups, ng = blockIdx.x - mg * ngroups;
  int mt = mg * WM + wm;
  int nt = ng * WN + wn;

  // per-lane pixel (column N = lane&15 of this wave's tile)
  int p = nt * 16 + (lane & 15);
  bool pvalid = p < NHW;
  int pp  = pvalid ? p : 0;
  int n   = pp / HW;
  int rem = pp - n * HW;
  int y = rem / W, x = rem - y * W;
  int hi = lane >> 4;                        // K-half of the fragment

  // incremental (r, cin0) for k = chunk*32 + hi*16
  int k0 = hi * 16;
  int r_ = k0 / Cp, cin0_ = k0 - r_ * Cp;
  const size_t baseN = (size_t)n * Cin * HW;

  v8f acc = {};
  const char* gA = (const char*)wpack + (size_t)mt * nkc * 1024
                   + (size_t)lane * 32;

  // ---- async-DMA one K-chunk of A (1 KB, contiguous) into LDS buffer b ----
  auto stageA = [&](int c, int b) {
    if (wn != 0) return;                     // wave-uniform: one wave per mt
    unsigned lds = (unsigned)(size_t)(void*)&sA[b][wm][0]
                   + (unsigned)lane * 32u;
    unsigned long long ga =
        (unsigned long long)(size_t)(gA + (size_t)c * 1024);
    asm volatile("global_load_async_to_lds_b128 %0, %1, off"
                 :: "v"(lds), "v"(ga) : "memory");
    asm volatile("global_load_async_to_lds_b128 %0, %1, off offset:16"
                 :: "v"(lds), "v"(ga) : "memory");
    __builtin_prefetch(gA + (size_t)(c + 1) * 1024, 0, 1);
  };

  // ---- gather one K-chunk of B into LDS buffer b ----
  auto stageB = [&](int b) {
    bool vr = (r_ < 9) && pvalid;
    int rc = r_ < 9 ? r_ : 8;
    int ky = rc / 3, kx = rc - (rc / 3) * 3;
    int iy = y + ky - 1, ix = x + kx - 1;
    bool vp = vr && (unsigned)iy < (unsigned)H && (unsigned)ix < (unsigned)W;
    int iyc = min(max(iy, 0), H - 1), ixc = min(max(ix, 0), W - 1);
    const float* bp = vin + baseN + (size_t)iyc * W + ixc;
    int cinS = cin0_ + wm * EPW;
    alignas(16) unsigned short vals[EPW];
#pragma unroll
    for (int e = 0; e < EPW; ++e) {
      int cin = cinS + e;
      int cc  = min(cin, Cin - 1);
      float ld = bp[(size_t)cc * HW];        // always in-bounds (clamped)
      vals[e] = f2bfu((vp && cin < Cin) ? ld : 0.f);
    }
    __builtin_memcpy(&sB[b][wn][lane * 16 + wm * EPW], vals, EPW * 2);
    cin0_ += 32;                              // advance to next chunk
    while (cin0_ >= Cp) { cin0_ -= Cp; ++r_; }
  };

  stageA(0, 0);
  stageB(0);
  for (int c = 0; c < nkc; ++c) {
    // issuing waves drain their async copies, then everyone syncs
    asm volatile("s_wait_asynccnt 0x0" ::: "memory");
    __syncthreads();
    v16bf afrag = *(const v16bf*)&sA[c & 1][wm][lane * 16];
    v16bf bfrag = *(const v16bf*)&sB[c & 1][wn][lane * 16];
    if (c + 1 < nkc) { stageA(c + 1, (c + 1) & 1); stageB((c + 1) & 1); }
    acc = __builtin_amdgcn_wmma_f32_16x16x32_bf16(
        false, afrag, false, bfrag, (short)0, acc, false, false);
  }

  if (pvalid) {
#pragma unroll
    for (int v = 0; v < 8; ++v) {             // C/D layout: VGPR v = row hi*8+v
      int m = mt * 16 + hi * 8 + v;
      out[((size_t)(n * Cout + m)) * HW + rem] = acc[v] + bias[m];
    }
  }
}

// ---- BN statistics: one block per channel, LDS tree reduction -------------
__global__ void k_bnstats(const float* __restrict__ v, float* __restrict__ mv,
                          int C, int HW) {
  int c = blockIdx.x;
  int NHW = 2 * HW;
  float s = 0.f, s2 = 0.f;
  for (int i = threadIdx.x; i < NHW; i += blockDim.x) {
    int n = i / HW, rem = i - n * HW;
    float val = v[((size_t)(n * C + c)) * HW + rem];
    s += val; s2 += val * val;
  }
  __shared__ float sh1[256], sh2[256];
  sh1[threadIdx.x] = s; sh2[threadIdx.x] = s2;
  __syncthreads();
  for (int o = 128; o > 0; o >>= 1) {
    if ((int)threadIdx.x < o) {
      sh1[threadIdx.x] += sh1[threadIdx.x + o];
      sh2[threadIdx.x] += sh2[threadIdx.x + o];
    }
    __syncthreads();
  }
  if (threadIdx.x == 0) {
    float m = sh1[0] / NHW;
    mv[c]     = m;
    mv[C + c] = fmaxf(sh2[0] / NHW - m * m, 0.f);
  }
}

__global__ void k_bnapply(const float* __restrict__ v, const float* __restrict__ mv,
                          const float* __restrict__ g, const float* __restrict__ b,
                          float* __restrict__ out, int C, int HW) {
  size_t total = (size_t)2 * C * HW;
  size_t i = (size_t)blockIdx.x * blockDim.x + threadIdx.x;
  if (i >= total) return;
  int c = (int)((i / HW) % C);
  out[i] = g[c] * (v[i] - mv[c]) * rsqrtf(mv[C + c] + 1e-5f) + b[c];
}

// ---- 2x2 max pool (tangent space) -----------------------------------------
__global__ void k_pool(const float* __restrict__ v, float* __restrict__ out,
                       int C, int H, int W) {
  int Ho = H / 2, Wo = W / 2;
  size_t total = (size_t)2 * C * Ho * Wo;
  size_t i = (size_t)blockIdx.x * blockDim.x + threadIdx.x;
  if (i >= total) return;
  int rem = (int)(i % (Ho * Wo));
  int t   = (int)(i / (Ho * Wo));
  int oy = rem / Wo, ox = rem - oy * Wo;
  const float* b = v + (size_t)t * H * W + (size_t)(oy * 2) * W + ox * 2;
  out[i] = fmaxf(fmaxf(b[0], b[1]), fmaxf(b[W], b[W + 1]));
}

// ---- 2x bilinear upsample, half-pixel centers -----------------------------
__global__ void k_up(const float* __restrict__ v, float* __restrict__ out,
                     int C, int H, int W) {
  int Ho = H * 2, Wo = W * 2;
  size_t total = (size_t)2 * C * Ho * Wo;
  size_t i = (size_t)blockIdx.x * blockDim.x + threadIdx.x;
  if (i >= total) return;
  int rem = (int)(i % (Ho * Wo));
  int t   = (int)(i / (Ho * Wo));
  int oy = rem / Wo, ox = rem - oy * Wo;
  float sy = (oy + 0.5f) * 0.5f - 0.5f;
  float sx = (ox + 0.5f) * 0.5f - 0.5f;
  int y0 = (int)floorf(sy), x0 = (int)floorf(sx);
  float fy = sy - y0, fx = sx - x0;
  int y1 = min(y0 + 1, H - 1), x1 = min(x0 + 1, W - 1);
  y0 = max(y0, 0); x0 = max(x0, 0);
  const float* b = v + (size_t)t * H * W;
  float v00 = b[y0 * W + x0], v01 = b[y0 * W + x1];
  float v10 = b[y1 * W + x0], v11 = b[y1 * W + x1];
  out[i] = (1.f - fy) * ((1.f - fx) * v00 + fx * v01) +
           fy * ((1.f - fx) * v10 + fx * v11);
}

// ---- beta-scaled channel concat (tangent space) ---------------------------
__global__ void k_cat2(const float* __restrict__ vx, const float* __restrict__ vy,
                       float* __restrict__ out, int Cx, int Cy, int HW,
                       float sx, float sy) {
  int C = Cx + Cy;
  size_t total = (size_t)2 * C * HW;
  size_t i = (size_t)blockIdx.x * blockDim.x + threadIdx.x;
  if (i >= total) return;
  int rem = (int)(i % HW);
  int t   = (int)(i / HW);
  int c = t % C, n = t / C;
  out[i] = (c < Cx)
      ? sx * vx[((size_t)(n * Cx + c)) * HW + rem]
      : sy * vy[((size_t)(n * Cy + (c - Cx))) * HW + rem];
}

// ---- final 1x1 conv (Cout=2) ----------------------------------------------
__global__ void k_conv1x1(const float* __restrict__ v, const float* __restrict__ W,
                          const float* __restrict__ b, float* __restrict__ out,
                          int Cin, int Cout, int HW) {
  size_t total = (size_t)2 * Cout * HW;
  size_t i = (size_t)blockIdx.x * blockDim.x + threadIdx.x;
  if (i >= total) return;
  int rem = (int)(i % HW);
  int t   = (int)(i / HW);
  int co = t % Cout, n = t / Cout;
  float s = b[co];
  const float* vp = v + (size_t)n * Cin * HW + rem;
  for (int ci = 0; ci < Cin; ++ci) s += W[co * Cin + ci] * vp[(size_t)ci * HW];
  out[i] = s;
}

// ===========================================================================
// Host orchestration
// ===========================================================================
namespace {

struct T { float* p; int C, H, W; };

float* g_persist; size_t g_poff, g_pcap;
float* g_scratch; size_t g_soff, g_scap;
hipStream_t g_s;
void* const* g_in;

// (in_ch, mid_ch, out_ch), indexed in setup_inputs() dict order
const int BC[15][3] = {
  {3,32,32},{32,64,64},{64,128,128},{128,256,256},{256,512,512},
  {96,32,32},{192,64,64},{384,128,128},{768,256,256},
  {128,32,32},{256,64,64},{512,128,128},
  {160,32,32},{320,64,64},
  {192,32,32}};

size_t elemsOf(const T& t) { return (size_t)2 * t.C * t.H * t.W; }

float* palloc(size_t e) {
  e = (e + 63) & ~(size_t)63;
  if (g_poff + e > g_pcap) g_poff = 0;          // wrap guard (ws too small)
  float* r = g_persist + g_poff; g_poff += e; return r;
}
float* salloc(size_t e) {
  e = (e + 63) & ~(size_t)63;
  if (g_soff + e > g_scap) g_soff = 0;          // wrap guard
  float* r = g_scratch + g_soff; g_soff += e; return r;
}

unsigned gsz(size_t total, int blk) { return (unsigned)((total + blk - 1) / blk); }

T logmapT(const T& x) {
  T o{salloc(elemsOf(x)), x.C, x.H, x.W};
  int HW = x.H * x.W;
  k_logmap<<<gsz(2 * HW, 256), 256, 0, g_s>>>(x.p, o.p, x.C, HW);
  return o;
}
T manifoldT(const T& x, int relu) {
  T o{salloc(elemsOf(x)), x.C, x.H, x.W};
  int HW = x.H * x.W;
  k_manifold<<<gsz(2 * HW, 256), 256, 0, g_s>>>(x.p, o.p, x.C, HW, relu);
  return o;
}
T convT(const T& xm, const float* W, const float* b, int Cout) {
  T t = logmapT(xm);
  int Cin = xm.C, H = xm.H, Wd = xm.W, HW = H * Wd;
  int Cp = (Cin + 15) & ~15;
  int nkc = (9 * Cp + 31) / 32;
  int mtiles = Cout / 16;
  size_t pel = (size_t)mtiles * nkc * 512;             // ushort elements
  unsigned short* wp = (unsigned short*)salloc(pel / 2 + 64);
  k_pack_w<<<1024, 256, 0, g_s>>>(W, wp, Cout, Cin);
  T y{salloc((size_t)2 * Cout * HW), Cout, H, Wd};
  int NHW = 2 * HW, ntiles = (NHW + 15) / 16;
  int WM = mtiles >= 8 ? 8 : mtiles;                   // 2, 4 or 8
  int WN = 8 / WM;
  unsigned blocks = (unsigned)((mtiles / WM) * ((ntiles + WN - 1) / WN));
  switch (WM) {
    case 2:
      k_conv3x3_wmma<2><<<blocks, 256, 0, g_s>>>(t.p, wp, b, y.p, Cin, Cout, H, Wd);
      break;
    case 4:
      k_conv3x3_wmma<4><<<blocks, 256, 0, g_s>>>(t.p, wp, b, y.p, Cin, Cout, H, Wd);
      break;
    default:
      k_conv3x3_wmma<8><<<blocks, 256, 0, g_s>>>(t.p, wp, b, y.p, Cin, Cout, H, Wd);
      break;
  }
  return manifoldT(y, 0);                              // expmap0(y + b)
}
T bnT(const T& xm, const float* g, const float* b) {
  T t = logmapT(xm);
  int HW = t.H * t.W;
  float* mv = salloc(2 * t.C);
  k_bnstats<<<t.C, 256, 0, g_s>>>(t.p, mv, t.C, HW);
  T o{salloc(elemsOf(t)), t.C, t.H, t.W};
  k_bnapply<<<gsz(elemsOf(t), 256), 256, 0, g_s>>>(t.p, mv, g, b, o.p, t.C, HW);
  return manifoldT(o, 0);
}
T reluT(const T& xm) { T t = logmapT(xm); return manifoldT(t, 1); }
T poolT(const T& xm) {
  T t = logmapT(xm);
  T o{salloc((size_t)2 * t.C * (t.H / 2) * (t.W / 2)), t.C, t.H / 2, t.W / 2};
  k_pool<<<gsz(elemsOf(o), 256), 256, 0, g_s>>>(t.p, o.p, t.C, t.H, t.W);
  return manifoldT(o, 0);
}
T upT(const T& xm) {
  T t = logmapT(xm);
  T o{salloc((size_t)2 * t.C * t.H * t.W * 4), t.C, t.H * 2, t.W * 2};
  k_up<<<gsz(elemsOf(o), 256), 256, 0, g_s>>>(t.p, o.p, t.C, t.H, t.W);
  return manifoldT(o, 0);
}
double betafn(double a, double b) {
  return exp(lgamma(a) + lgamma(b) - lgamma(a + b));
}
T pcatT(const T& x, const T& y) {
  T tx = logmapT(x), ty = logmapT(y);
  double bn = betafn((x.C + y.C) / 2.0, 0.5);
  float sx = (float)(bn / betafn(x.C / 2.0, 0.5));
  float sy = (float)(bn / betafn(y.C / 2.0, 0.5));
  T o{salloc((size_t)2 * (x.C + y.C) * x.H * x.W), x.C + y.C, x.H, x.W};
  int HW = x.H * x.W;
  k_cat2<<<gsz(elemsOf(o), 256), 256, 0, g_s>>>(tx.p, ty.p, o.p, x.C, y.C, HW,
                                                sx, sy);
  return manifoldT(o, 0);
}
T vggT(const T& x, int bi) {
  const float* W1  = (const float*)g_in[1 + 8 * bi + 0];
  const float* b1  = (const float*)g_in[1 + 8 * bi + 1];
  const float* g1  = (const float*)g_in[1 + 8 * bi + 2];
  const float* be1 = (const float*)g_in[1 + 8 * bi + 3];
  const float* W2  = (const float*)g_in[1 + 8 * bi + 4];
  const float* b2  = (const float*)g_in[1 + 8 * bi + 5];
  const float* g2  = (const float*)g_in[1 + 8 * bi + 6];
  const float* be2 = (const float*)g_in[1 + 8 * bi + 7];
  T a = convT(x, W1, b1, BC[bi][1]);
  a = bnT(a, g1, be1);
  a = reluT(a);
  a = convT(a, W2, b2, BC[bi][2]);
  a = bnT(a, g2, be2);
  a = reluT(a);
  return a;
}
T persistT(const T& t) {
  size_t e = elemsOf(t);
  T o{palloc(e), t.C, t.H, t.W};
  hipMemcpyAsync(o.p, t.p, e * sizeof(float), hipMemcpyDeviceToDevice, g_s);
  return o;
}

} // namespace

extern "C" void kernel_launch(void* const* d_in, const int* in_sizes, int n_in,
                              void* d_out, int out_size, void* d_ws, size_t ws_size,
                              hipStream_t stream) {
  (void)in_sizes; (void)n_in; (void)out_size;
  g_s = stream; g_in = d_in;
  size_t wf = ws_size / sizeof(float);
  g_persist = (float*)d_ws;  g_pcap = wf / 4;        g_poff = 0;
  g_scratch = g_persist + g_pcap; g_scap = wf - g_pcap; g_soff = 0;

  T xraw{(float*)d_in[0], 3, 256, 256};
  T xin = persistT(manifoldT(xraw, 0));                       g_soff = 0;
  T x0_0 = persistT(vggT(xin, 0));                            g_soff = 0;
  T x1_0 = persistT(vggT(poolT(x0_0), 1));                    g_soff = 0;
  T x0_1 = persistT(vggT(pcatT(x0_0, upT(x1_0)), 5));         g_soff = 0;
  T x2_0 = persistT(vggT(poolT(x1_0), 2));                    g_soff = 0;
  T x1_1 = persistT(vggT(pcatT(x1_0, upT(x2_0)), 6));         g_soff = 0;
  T x0_2 = persistT(vggT(pcatT(x0_0, pcatT(x0_1, upT(x1_1))), 9));      g_soff = 0;
  T x3_0 = persistT(vggT(poolT(x2_0), 3));                    g_soff = 0;
  T x2_1 = persistT(vggT(pcatT(x2_0, upT(x3_0)), 7));         g_soff = 0;
  T x1_2 = persistT(vggT(pcatT(x1_0, pcatT(x1_1, upT(x2_1))), 10));     g_soff = 0;
  T x0_3 = persistT(vggT(pcatT(x0_0, pcatT(x0_1, pcatT(x0_2, upT(x1_2)))), 12));
                                                              g_soff = 0;
  T x4_0 = persistT(vggT(poolT(x3_0), 4));                    g_soff = 0;
  T x3_1 = persistT(vggT(pcatT(x3_0, upT(x4_0)), 8));         g_soff = 0;
  T x2_2 = persistT(vggT(pcatT(x2_0, pcatT(x2_1, upT(x3_1))), 11));     g_soff = 0;
  T x1_3 = persistT(vggT(pcatT(x1_0, pcatT(x1_1, pcatT(x1_2, upT(x2_2)))), 13));
                                                              g_soff = 0;
  T x0_4 = vggT(pcatT(x0_0, pcatT(x0_1, pcatT(x0_2, pcatT(x0_3, upT(x1_3))))), 14);

  // final 1x1 hconv + logmap0 to output
  const int HW0 = 256 * 256;
  T tf = logmapT(x0_4);
  float* yf = salloc((size_t)2 * 2 * HW0);
  k_conv1x1<<<gsz((size_t)2 * 2 * HW0, 256), 256, 0, stream>>>(
      tf.p, (const float*)d_in[121], (const float*)d_in[122], yf, 32, 2, HW0);
  T ym{yf, 2, 256, 256};
  T mf = manifoldT(ym, 0);
  k_logmap<<<gsz(2 * HW0, 256), 256, 0, stream>>>(mf.p, (float*)d_out, 2, HW0);
}
